// PixelContrastiveLoss_88038239634215
// MI455X (gfx1250) — compile-verified, matrix operand order checked by reading, not run on hardware
//
#include <hip/hip_runtime.h>

typedef __attribute__((ext_vector_type(16))) _Float16 v16h;
typedef __attribute__((ext_vector_type(8)))  _Float16 v8h;
typedef __attribute__((ext_vector_type(8)))  float    v8f;

#define B_IMG   16
#define C_DIM   256
#define HW_PIX  16384
#define KSMP    1024
#define TWOK    2048
#define NTILE   128         // 2048 / 16 column tiles
#define IGNORE_L 255
#define INV_T   (1.0f / 0.07f)
#define EPS_L   1e-8f

// ---------------------------------------------------------------------------
// Fragment load for 16-bit WMMA A/B operands (16x32 per k-step).
// lane L: M (or N) = L & 15 ; K base = (L < 16 ? 0 : 8);
// halves 0..7  -> K = base + h        (contiguous 8 halfs = 16B)
// halves 8..15 -> K = base + 16 + h-8 (contiguous 8 halfs = 16B)
// ---------------------------------------------------------------------------
__device__ __forceinline__ v16h load_frag(const _Float16* row, int kk, int halfOff) {
    const _Float16* p = row + kk * 32 + halfOff;
    v8h lo = *(const v8h*)(p);
    v8h hi = *(const v8h*)(p + 16);
    return __builtin_shufflevector(lo, hi, 0,1,2,3,4,5,6,7,8,9,10,11,12,13,14,15);
}

// ---------------------------------------------------------------------------
// Kernel 1: per-image stable sampler (one wave per image).
// Positives (label != 0 && != 255) in raster order first, then background,
// padded with idx=-1 / label=255.
// ---------------------------------------------------------------------------
__global__ void sample_kernel(const int* __restrict__ labels,
                              int* __restrict__ selIdx,
                              int* __restrict__ selLab,
                              int* __restrict__ bgBuf) {
    const int b    = blockIdx.x;
    const int lane = threadIdx.x;
    const int* L = labels + b * HW_PIX;
    const unsigned long long lmask = (1ull << lane) - 1ull;

    int fg = 0, bg = 0;
    for (int i0 = 0; i0 < HW_PIX; i0 += 32) {
        const int i   = i0 + lane;
        const int lab = L[i];
        const bool isFg = (lab != 0) && (lab != IGNORE_L);
        const bool isBg = (lab == 0);
        const unsigned long long mFg = __ballot(isFg);
        const unsigned long long mBg = __ballot(isBg);
        if (isFg) {
            const int p = fg + __popcll(mFg & lmask);
            if (p < KSMP) { selIdx[b * KSMP + p] = i; selLab[b * KSMP + p] = lab; }
        }
        if (isBg) {
            const int p = bg + __popcll(mBg & lmask);
            if (p < KSMP) bgBuf[b * KSMP + p] = i;
        }
        fg += __popcll(mFg);
        bg += __popcll(mBg);
    }
    // make earlier stores visible before cross-lane readback
    asm volatile("s_wait_storecnt 0x0" ::: "memory");
    for (int k = lane; k < KSMP; k += 32) {
        if (k >= fg) {
            const int t = k - fg;
            if (t < bg && t < KSMP) {
                selIdx[b * KSMP + k] = bgBuf[b * KSMP + t];
                selLab[b * KSMP + k] = 0;
            } else {
                selIdx[b * KSMP + k] = -1;
                selLab[b * KSMP + k] = IGNORE_L;
            }
        }
    }
}

// ---------------------------------------------------------------------------
// Kernel 2: gather sampled feature vectors (strided by HW in [B,C,H,W]),
// L2-normalize, store fp16 row-major [B][2K][C].  One wave per sample.
// ---------------------------------------------------------------------------
__global__ void gather_kernel(const float* __restrict__ feats,
                              const int* __restrict__ selIdx,
                              const int* __restrict__ selLab,
                              _Float16* __restrict__ Fh,
                              int* __restrict__ lp) {
    const int wave = blockIdx.x * 8 + (threadIdx.x >> 5);
    const int lane = threadIdx.x & 31;
    const int b  = wave / TWOK;
    const int k2 = wave % TWOK;
    const int srcB = (k2 < KSMP) ? b : (b + 1) % B_IMG;
    const int k    = k2 & (KSMP - 1);
    const int idx  = selIdx[srcB * KSMP + k];
    const int lab  = selLab[srcB * KSMP + k];

    float v[8];
    if (idx < 0) {
        #pragma unroll
        for (int c8 = 0; c8 < 8; ++c8) v[c8] = 0.f;
    } else {
        const float* p = feats + (size_t)srcB * C_DIM * HW_PIX + idx;
        #pragma unroll
        for (int c8 = 0; c8 < 8; ++c8)
            v[c8] = p[(size_t)(c8 * 32 + lane) * HW_PIX];
    }
    float s = 0.f;
    #pragma unroll
    for (int c8 = 0; c8 < 8; ++c8) s += v[c8] * v[c8];
    #pragma unroll
    for (int off = 16; off > 0; off >>= 1) s += __shfl_xor(s, off, 32);
    const float inv = 1.f / fmaxf(sqrtf(s), 1e-12f);

    _Float16* row = Fh + (size_t)(b * TWOK + k2) * C_DIM;
    #pragma unroll
    for (int c8 = 0; c8 < 8; ++c8)
        row[c8 * 32 + lane] = (_Float16)(v[c8] * inv);
    if (lane == 0) lp[b * TWOK + k2] = lab;
}

// ---------------------------------------------------------------------------
// Kernel 3: fused contrastive loss over one 16-row tile of one image.
// Three WMMA sweeps over the 128 column tiles:
//   pass 1: row max  (clamp-init 0: diagonal gives 1/T for valid rows,
//           invalid rows are all-zero -> max 0, identical to reference)
//   pass 2: sum_neg (same-K-block only -> skip half the tiles)
//   pass 3: positive log-prob sum + positive count
// ---------------------------------------------------------------------------
__global__ void __launch_bounds__(256)
contrast_main(const _Float16* __restrict__ Fh,
              const int* __restrict__ lp,
              float* __restrict__ lossSum,
              float* __restrict__ anchorCnt) {
    const int b    = blockIdx.x >> 7;    // 128 row tiles per image
    const int rt   = blockIdx.x & 127;
    const int lane = threadIdx.x & 31;
    const int wave = threadIdx.x >> 5;   // 0..7
    const int half = (lane >> 4) & 1;
    const int lrow = lane & 15;
    const int halfOff = half * 8;
    const int rowbase = rt * 16;

    const _Float16* base = Fh + (size_t)b * TWOK * C_DIM;
    const int* lpb = lp + b * TWOK;

    __shared__ int   sMaxBits[16];
    __shared__ float sNeg[16];
    __shared__ float sPos[16];
    __shared__ float sCnt[16];
    if (threadIdx.x < 16) {
        sMaxBits[threadIdx.x] = 0;       // float 0.0f
        sNeg[threadIdx.x] = 0.f;
        sPos[threadIdx.x] = 0.f;
        sCnt[threadIdx.x] = 0.f;
    }
    __syncthreads();

    // preload A fragments (this block's 16 rows, all 8 k-steps)
    v16h a[8];
    const _Float16* arow = base + (size_t)(rowbase + lrow) * C_DIM;
    #pragma unroll
    for (int kk = 0; kk < 8; ++kk) a[kk] = load_frag(arow, kk, halfOff);

    int  labRow[8];
    bool vRow[8];
    #pragma unroll
    for (int r = 0; r < 8; ++r) {
        labRow[r] = lpb[rowbase + r + 8 * half];
        vRow[r]   = (labRow[r] != IGNORE_L);
    }

    // ---------------- pass 1: row max ----------------
    float mx[8];
    #pragma unroll
    for (int r = 0; r < 8; ++r) mx[r] = 0.f;
    for (int jt = wave; jt < NTILE; jt += 8) {
        const _Float16* brow = base + (size_t)(jt * 16 + lrow) * C_DIM;
        if (jt + 8 < NTILE)
            __builtin_prefetch(base + (size_t)((jt + 8) * 16 + lrow) * C_DIM, 0, 1);
        v8f c = {0.f, 0.f, 0.f, 0.f, 0.f, 0.f, 0.f, 0.f};
        #pragma unroll
        for (int kk = 0; kk < 8; ++kk) {
            v16h bf = load_frag(brow, kk, halfOff);
            c = __builtin_amdgcn_wmma_f32_16x16x32_f16(false, a[kk], false, bf,
                                                       (short)0, c, false, false);
        }
        #pragma unroll
        for (int r = 0; r < 8; ++r) mx[r] = fmaxf(mx[r], c[r] * INV_T);
    }
    #pragma unroll
    for (int off = 8; off > 0; off >>= 1)
        #pragma unroll
        for (int r = 0; r < 8; ++r) mx[r] = fmaxf(mx[r], __shfl_xor(mx[r], off, 16));
    if (lrow == 0) {
        #pragma unroll
        for (int r = 0; r < 8; ++r)
            atomicMax(&sMaxBits[r + 8 * half], __float_as_int(mx[r]));
    }
    __syncthreads();
    float rm[8];
    #pragma unroll
    for (int r = 0; r < 8; ++r) rm[r] = __int_as_float(sMaxBits[r + 8 * half]);

    // ---------------- pass 2: sum of exp over negatives (same K-block) ------
    const bool rowBlk = (rt < NTILE / 2);
    float sn[8];
    #pragma unroll
    for (int r = 0; r < 8; ++r) sn[r] = 0.f;
    for (int jt = wave; jt < NTILE; jt += 8) {
        if ((jt < NTILE / 2) != rowBlk) continue;   // same_img mask
        const int  labC = lpb[jt * 16 + lrow];
        const bool vC   = (labC != IGNORE_L);
        const _Float16* brow = base + (size_t)(jt * 16 + lrow) * C_DIM;
        v8f c = {0.f, 0.f, 0.f, 0.f, 0.f, 0.f, 0.f, 0.f};
        #pragma unroll
        for (int kk = 0; kk < 8; ++kk) {
            v16h bf = load_frag(brow, kk, halfOff);
            c = __builtin_amdgcn_wmma_f32_16x16x32_f16(false, a[kk], false, bf,
                                                       (short)0, c, false, false);
        }
        #pragma unroll
        for (int r = 0; r < 8; ++r) {
            if (vRow[r] && vC && (labRow[r] != labC))
                sn[r] += __expf(c[r] * INV_T - rm[r]);
        }
    }
    #pragma unroll
    for (int off = 8; off > 0; off >>= 1)
        #pragma unroll
        for (int r = 0; r < 8; ++r) sn[r] += __shfl_xor(sn[r], off, 16);
    if (lrow == 0) {
        #pragma unroll
        for (int r = 0; r < 8; ++r) atomicAdd(&sNeg[r + 8 * half], sn[r]);
    }
    __syncthreads();
    float snv[8];
    #pragma unroll
    for (int r = 0; r < 8; ++r) snv[r] = sNeg[r + 8 * half];

    // ---------------- pass 3: positive log-prob sum + count -----------------
    float ps[8], pc[8];
    #pragma unroll
    for (int r = 0; r < 8; ++r) { ps[r] = 0.f; pc[r] = 0.f; }
    for (int jt = wave; jt < NTILE; jt += 8) {
        const int  colIdx = jt * 16 + lrow;
        const int  labC = lpb[colIdx];
        const bool vC   = (labC != IGNORE_L);
        const _Float16* brow = base + (size_t)colIdx * C_DIM;
        v8f c = {0.f, 0.f, 0.f, 0.f, 0.f, 0.f, 0.f, 0.f};
        #pragma unroll
        for (int kk = 0; kk < 8; ++kk) {
            v16h bf = load_frag(brow, kk, halfOff);
            c = __builtin_amdgcn_wmma_f32_16x16x32_f16(false, a[kk], false, bf,
                                                       (short)0, c, false, false);
        }
        #pragma unroll
        for (int r = 0; r < 8; ++r) {
            const int rowIdx = rowbase + r + 8 * half;
            if (vRow[r] && vC && (labRow[r] == labC) && (rowIdx != colIdx)) {
                const float t = c[r] * INV_T - rm[r];
                ps[r] += t - __logf(__expf(t) + snv[r] + EPS_L);
                pc[r] += 1.f;
            }
        }
    }
    #pragma unroll
    for (int off = 8; off > 0; off >>= 1)
        #pragma unroll
        for (int r = 0; r < 8; ++r) {
            ps[r] += __shfl_xor(ps[r], off, 16);
            pc[r] += __shfl_xor(pc[r], off, 16);
        }
    if (lrow == 0) {
        #pragma unroll
        for (int r = 0; r < 8; ++r) {
            atomicAdd(&sPos[r + 8 * half], ps[r]);
            atomicAdd(&sCnt[r + 8 * half], pc[r]);
        }
    }
    __syncthreads();

    if (threadIdx.x < 16) {
        const float cnt  = sCnt[threadIdx.x];
        const float mean = sPos[threadIdx.x] / fmaxf(cnt, 1.f);
        const float va   = (cnt > 0.f) ? 1.f : 0.f;
        atomicAdd(&lossSum[b], -mean * va);
        atomicAdd(&anchorCnt[b], va);
    }
}

// ---------------------------------------------------------------------------
__global__ void init_kernel(float* lossSum, float* anchorCnt) {
    if (threadIdx.x < B_IMG) { lossSum[threadIdx.x] = 0.f; anchorCnt[threadIdx.x] = 0.f; }
}

__global__ void finalize_kernel(const float* __restrict__ lossSum,
                                const float* __restrict__ anchorCnt,
                                float* __restrict__ out) {
    if (threadIdx.x == 0) {
        float s = 0.f;
        for (int b = 0; b < B_IMG; ++b)
            s += lossSum[b] / fmaxf(anchorCnt[b], 1.f);
        out[0] = s / (float)B_IMG;
    }
}

// ---------------------------------------------------------------------------
extern "C" void kernel_launch(void* const* d_in, const int* in_sizes, int n_in,
                              void* d_out, int out_size, void* d_ws, size_t ws_size,
                              hipStream_t stream) {
    const float* feats  = (const float*)d_in[0];
    const int*   labels = (const int*)d_in[1];
    float*       out    = (float*)d_out;

    char* w = (char*)d_ws;
    size_t off = 0;
    int* selIdx = (int*)(w + off); off += (size_t)B_IMG * KSMP * sizeof(int);
    int* selLab = (int*)(w + off); off += (size_t)B_IMG * KSMP * sizeof(int);
    int* bgBuf  = (int*)(w + off); off += (size_t)B_IMG * KSMP * sizeof(int);
    int* lpBuf  = (int*)(w + off); off += (size_t)B_IMG * TWOK * sizeof(int);
    float* lossSum   = (float*)(w + off); off += 64;
    float* anchorCnt = (float*)(w + off); off += 64;
    off = (off + 511) & ~(size_t)511;
    _Float16* Fh = (_Float16*)(w + off);   // 16 MB: [B][2K][C] fp16

    init_kernel<<<1, 64, 0, stream>>>(lossSum, anchorCnt);
    sample_kernel<<<B_IMG, 32, 0, stream>>>(labels, selIdx, selLab, bgBuf);
    gather_kernel<<<(B_IMG * TWOK) / 8, 256, 0, stream>>>(feats, selIdx, selLab, Fh, lpBuf);
    contrast_main<<<B_IMG * NTILE, 256, 0, stream>>>(Fh, lpBuf, lossSum, anchorCnt);
    finalize_kernel<<<1, 32, 0, stream>>>(lossSum, anchorCnt, out);
}
